// AttenionDecoder_49555332661469
// MI455X (gfx1250) — matrix operand order repaired and checked by reading, compiled, not run on hardware
//
#include <hip/hip_runtime.h>
#include <hip/hip_bf16.h>

#define B_   32
#define S_   256
#define E_   512
#define U_   512
#define V_   8000
#define T_   64
#define PAD_ 2

typedef __attribute__((ext_vector_type(16))) __bf16 v16bf;
typedef __attribute__((ext_vector_type(8)))  float  v8f;

union ABPack { v16bf v; uint4 q[2]; };

__device__ __forceinline__ float sigm(float x) { return 1.0f / (1.0f + __expf(-x)); }

__device__ __forceinline__ float ftanh(float x) {
  float ax = fminf(fabsf(x), 15.0f);
  float e  = __expf(2.0f * ax);
  float r  = (e - 1.0f) / (e + 1.0f);
  return copysignf(r, x);
}

// One 16x16 (M x N) fp32 accumulator tile of A(16 x K, row-major, stride lda)
// times Bt (N-major: Bt[n][k], stride K), K multiple of 32, all bf16.
// A layout per lane: lane<16 -> M=lane, K 0..15; lane>=16 -> same M, K 16..31
// (contiguous, matching B).
// B layout: lane%16 = column N, lane/16 selects K half (0..15 / 16..31).
__device__ __forceinline__ v8f wmma_dot(const __bf16* __restrict__ A, int lda,
                                        const __bf16* __restrict__ Bt, int K,
                                        int lane) {
  v8f acc = {};
  const int mr   = lane & 15;
  const int half = lane >> 4;
  const __bf16* arow = A  + (long long)mr * lda + half * 16;
  const __bf16* bcol = Bt + (long long)mr * K   + half * 16;
  for (int k0 = 0; k0 < K; k0 += 32) {
    ABPack a, b;
    a.q[0] = *(const uint4*)(arow + k0);
    a.q[1] = *(const uint4*)(arow + k0 + 8);
    b.q[0] = *(const uint4*)(bcol + k0);
    b.q[1] = *(const uint4*)(bcol + k0 + 8);
    if (k0 + 128 < K) __builtin_prefetch(bcol + k0 + 128, 0, 1);
    acc = __builtin_amdgcn_wmma_f32_16x16x32_bf16(false, a.v, false, b.v,
                                                  (short)0, acc, false, false);
  }
  return acc;
}

// ---- setup kernels ---------------------------------------------------------

// src fp32 [K][N] row-major -> dst bf16 [N][K]
__global__ void k_tconv(const float* __restrict__ src, __bf16* __restrict__ dst,
                        int K, int N) {
  long long idx   = (long long)blockIdx.x * 256 + threadIdx.x;
  long long total = (long long)K * N;
  if (idx >= total) return;
  int n = (int)(idx / K);
  int k = (int)(idx % K);
  dst[idx] = (__bf16)src[(long long)k * N + n];
}

__global__ void k_cvt(const float* __restrict__ src, __bf16* __restrict__ dst,
                      int total) {
  int idx = blockIdx.x * 256 + threadIdx.x;
  if (idx < total) dst[idx] = (__bf16)src[idx];
}

__global__ void k_init(__bf16* ah, __bf16* hc, float* c,
                       const int* __restrict__ labels, int* gt) {
  int idx = blockIdx.x * 256 + threadIdx.x;
  if (idx < B_ * 1024) { ah[idx] = (__bf16)0.0f; hc[idx] = (__bf16)0.0f; }
  if (idx < B_ * U_) c[idx] = 0.0f;
  if (idx < B_) {
    int cnt = 0;
    for (int t = 0; t < T_; ++t) cnt += (labels[idx * T_ + t] == PAD_) ? 1 : 0;
    gt[idx] = T_ - cnt;
  }
}

// keys = inputs_bf(8192x512) @ memory_w -> fp32 [8192][512]
__global__ void k_keys(const __bf16* __restrict__ A, const __bf16* __restrict__ Bt,
                       float* __restrict__ out) {
  int wave = threadIdx.x >> 5, lane = threadIdx.x & 31;
  int bm = blockIdx.x >> 3, bn = blockIdx.x & 7;
  int m0 = bm * 32 + (wave >> 2) * 16;
  int n0 = bn * 64 + (wave & 3) * 16;
  v8f acc = wmma_dot(A + (long long)m0 * 512, 512, Bt + (long long)n0 * 512, 512, lane);
  int n = n0 + (lane & 15);
  int mbase = m0 + ((lane >> 4) << 3);
#pragma unroll
  for (int r = 0; r < 8; ++r)
    out[(long long)(mbase + r) * 512 + n] = acc[r];
}

// ---- per-step kernels ------------------------------------------------------

// Fused LSTM: z = [a,h] @ W2 + lstm_w[label] + b ; gates -> h,c.
// 32 WGs, each handles 16 columns of U across all 4 gates (8 waves = 2M x 4gates).
__global__ void k_lstm(const __bf16* __restrict__ ah, const __bf16* __restrict__ Wt,
                       const float* __restrict__ lstm_w, const float* __restrict__ lstm_b,
                       const int* __restrict__ labels, int t,
                       float* __restrict__ c_state, __bf16* __restrict__ hc) {
  __shared__ float zl[4][2][8][32];
  int wave = threadIdx.x >> 5, lane = threadIdx.x & 31;
  int gate = wave & 3, mt = wave >> 2;
  int ncol0 = blockIdx.x * 16;
  int gcol0 = gate * 512 + ncol0;

  v8f acc = wmma_dot(ah + (long long)mt * 16 * 1024, 1024,
                     Wt + (long long)gcol0 * 1024, 1024, lane);

  int n = lane & 15;
  int mbase = mt * 16 + ((lane >> 4) << 3);
  int col = gcol0 + n;
  float bv = lstm_b[col];
#pragma unroll
  for (int r = 0; r < 8; ++r) {
    int m = mbase + r;
    int lab = labels[m * T_ + t];
    zl[gate][mt][r][lane] = acc[r] + bv + lstm_w[(long long)lab * 2048 + col];
  }
  __syncthreads();

  for (int e = threadIdx.x; e < 512; e += 256) {
    int m = e >> 4, nn = e & 15;
    int mt2 = m >> 4, mr = m & 15;
    int r = mr & 7, ln = nn + ((mr >> 3) << 4);
    float zi = zl[0][mt2][r][ln];
    float zf = zl[1][mt2][r][ln];
    float zg = zl[2][mt2][r][ln];
    float zo = zl[3][mt2][r][ln];
    int uc = ncol0 + nn;
    float cold = c_state[m * U_ + uc];
    float cn = sigm(zf) * cold + sigm(zi) * ftanh(zg);
    float h  = sigm(zo) * ftanh(cn);
    c_state[m * U_ + uc] = cn;
    hc[m * 1024 + uc] = (__bf16)h;   // h half of [h, ctx]
  }
}

// q = h @ query_w -> fp32 [32][512]; also republish h into ah[:,512:1024]
__global__ void k_query(const __bf16* __restrict__ hc, const __bf16* __restrict__ Bt,
                        float* __restrict__ qout, __bf16* __restrict__ ah) {
  int wave = threadIdx.x >> 5, lane = threadIdx.x & 31;
  int m0 = (wave >> 2) * 16;
  int n0 = blockIdx.x * 64 + (wave & 3) * 16;
  v8f acc = wmma_dot(hc + (long long)m0 * 1024, 1024, Bt + (long long)n0 * 512, 512, lane);
  int n = n0 + (lane & 15);
  int mbase = m0 + ((lane >> 4) << 3);
#pragma unroll
  for (int r = 0; r < 8; ++r) {
    int m = mbase + r;
    qout[m * U_ + n] = acc[r];
    ah[(long long)m * 1024 + 512 + n] = hc[(long long)m * 1024 + n];
  }
}

// scores + softmax + context; one WG per batch row.
__global__ void k_score(const float* __restrict__ keys, const float* __restrict__ q,
                        const float* __restrict__ av, const float* __restrict__ values,
                        __bf16* __restrict__ hc) {
  __shared__ float qs[512], vs[512], red[256], al[256];
  int b = blockIdx.x, tid = threadIdx.x;
  qs[tid] = q[b * U_ + tid];        qs[tid + 256] = q[b * U_ + tid + 256];
  vs[tid] = av[tid];                vs[tid + 256] = av[tid + 256];
  __syncthreads();

  const float* krow = keys + ((long long)b * S_ + tid) * U_;
  float acc = 0.0f;
#pragma unroll 4
  for (int u = 0; u < U_; ++u) acc += vs[u] * ftanh(krow[u] + qs[u]);

  red[tid] = acc; __syncthreads();
  for (int off = 128; off > 0; off >>= 1) {
    if (tid < off) red[tid] = fmaxf(red[tid], red[tid + off]);
    __syncthreads();
  }
  float mx = red[0]; __syncthreads();
  float ex = __expf(acc - mx);
  red[tid] = ex; __syncthreads();
  for (int off = 128; off > 0; off >>= 1) {
    if (tid < off) red[tid] += red[tid + off];
    __syncthreads();
  }
  float inv = 1.0f / red[0];
  al[tid] = ex * inv; __syncthreads();

  const float* vb = values + (long long)b * S_ * E_;
  float c0 = 0.0f, c1 = 0.0f;
  for (int s = 0; s < S_; ++s) {
    float a = al[s];
    const float* vr = vb + (long long)s * E_;
    c0 += a * vr[tid];
    c1 += a * vr[tid + 256];
  }
  hc[(long long)b * 1024 + 512 + tid]       = (__bf16)c0;
  hc[(long long)b * 1024 + 512 + 256 + tid] = (__bf16)c1;
}

// a = [h,ctx] @ attention_w -> bf16 into ah[:,0:512]
__global__ void k_attn(const __bf16* __restrict__ hc, const __bf16* __restrict__ Bt,
                       __bf16* __restrict__ ah) {
  int wave = threadIdx.x >> 5, lane = threadIdx.x & 31;
  int m0 = (wave >> 2) * 16;
  int n0 = blockIdx.x * 64 + (wave & 3) * 16;
  v8f acc = wmma_dot(hc + (long long)m0 * 1024, 1024, Bt + (long long)n0 * 1024, 1024, lane);
  int n = n0 + (lane & 15);
  int mbase = m0 + ((lane >> 4) << 3);
#pragma unroll
  for (int r = 0; r < 8; ++r)
    ah[(long long)(mbase + r) * 1024 + n] = (__bf16)acc[r];
}

// logits = a @ project_w + b, masked by t < gt_len
__global__ void k_proj(const __bf16* __restrict__ ah, const __bf16* __restrict__ Bt,
                       const float* __restrict__ bias, const int* __restrict__ gt,
                       int t, float* __restrict__ out) {
  int wave = threadIdx.x >> 5, lane = threadIdx.x & 31;
  int m0 = (wave >> 2) * 16;
  int n0 = blockIdx.x * 64 + (wave & 3) * 16;
  v8f acc = wmma_dot(ah + (long long)m0 * 1024, 1024, Bt + (long long)n0 * 512, 512, lane);
  int n = n0 + (lane & 15);
  int mbase = m0 + ((lane >> 4) << 3);
  float bv = bias[n];
#pragma unroll
  for (int r = 0; r < 8; ++r) {
    int m = mbase + r;
    float v = (t < gt[m]) ? (acc[r] + bv) : 0.0f;
    out[((long long)m * T_ + t) * V_ + n] = v;
  }
}

// ---- launch ----------------------------------------------------------------

extern "C" void kernel_launch(void* const* d_in, const int* in_sizes, int n_in,
                              void* d_out, int out_size, void* d_ws, size_t ws_size,
                              hipStream_t stream) {
  (void)in_sizes; (void)n_in; (void)out_size; (void)ws_size;
  const float* inputs   = (const float*)d_in[0];
  const int*   labels   = (const int*)  d_in[1];
  const float* memory_w = (const float*)d_in[2];
  const float* lstm_w   = (const float*)d_in[3];
  const float* lstm_b   = (const float*)d_in[4];
  const float* query_w  = (const float*)d_in[5];
  const float* attn_v   = (const float*)d_in[6];
  const float* attn_w   = (const float*)d_in[7];
  const float* proj_w   = (const float*)d_in[8];
  const float* proj_b   = (const float*)d_in[9];
  float* out = (float*)d_out;

  char* ws = (char*)d_ws;
  size_t o = 0;
  auto take = [&](size_t bytes) -> char* {
    char* p = ws + o;
    o += (bytes + 255) & ~(size_t)255;
    return p;
  };
  __bf16* lstm_wt  = (__bf16*)take((size_t)2048 * 1024 * 2);
  __bf16* proj_wt  = (__bf16*)take((size_t)8000 * 512 * 2);
  __bf16* attn_wt  = (__bf16*)take((size_t)512 * 1024 * 2);
  __bf16* query_wt = (__bf16*)take((size_t)512 * 512 * 2);
  __bf16* mem_wt   = (__bf16*)take((size_t)512 * 512 * 2);
  __bf16* in_bf    = (__bf16*)take((size_t)8192 * 512 * 2);
  float*  keys     = (float*) take((size_t)8192 * 512 * 4);
  float*  qbuf     = (float*) take((size_t)32 * 512 * 4);
  float*  cst      = (float*) take((size_t)32 * 512 * 4);
  __bf16* ah       = (__bf16*)take((size_t)32 * 1024 * 2);
  __bf16* hc       = (__bf16*)take((size_t)32 * 1024 * 2);
  int*    gt       = (int*)   take((size_t)32 * 4);

  dim3 blk(256);
  k_tconv<<<dim3((2048 * 1024 + 255) / 256), blk, 0, stream>>>(lstm_w + (size_t)V_ * 2048, lstm_wt, 1024, 2048);
  k_tconv<<<dim3((8000 * 512 + 255) / 256),  blk, 0, stream>>>(proj_w,  proj_wt,  512, 8000);
  k_tconv<<<dim3((512 * 1024 + 255) / 256),  blk, 0, stream>>>(attn_w,  attn_wt,  1024, 512);
  k_tconv<<<dim3((512 * 512 + 255) / 256),   blk, 0, stream>>>(query_w, query_wt, 512, 512);
  k_tconv<<<dim3((512 * 512 + 255) / 256),   blk, 0, stream>>>(memory_w, mem_wt,  512, 512);
  k_cvt  <<<dim3((8192 * 512) / 256),        blk, 0, stream>>>(inputs, in_bf, 8192 * 512);
  k_init <<<dim3(128),                       blk, 0, stream>>>(ah, hc, cst, labels, gt);
  k_keys <<<dim3(2048),                      blk, 0, stream>>>(in_bf, mem_wt, keys);

  for (int t = 0; t < T_; ++t) {
    k_lstm <<<dim3(32),  blk, 0, stream>>>(ah, lstm_wt, lstm_w, lstm_b, labels, t, cst, hc);
    k_query<<<dim3(8),   blk, 0, stream>>>(hc, query_wt, qbuf, ah);
    k_score<<<dim3(32),  blk, 0, stream>>>(keys, qbuf, attn_v, inputs, hc);
    k_attn <<<dim3(8),   blk, 0, stream>>>(hc, attn_wt, ah);
    k_proj <<<dim3(125), blk, 0, stream>>>(ah, proj_wt, proj_b, gt, t, out);
  }
}
